// AdaMEOW_12515534700965
// MI455X (gfx1250) — compile-verified, matrix operand order checked by reading, not run on hardware
//
#include <hip/hip_runtime.h>
#include <hip/hip_bf16.h>
#include <math.h>

// Problem dims (from reference)
#define N_NODES 1024
#define M_NODES 4096
#define D0_DIM  1024
#define D1_DIM  512
#define H_DIM   512
#define E_DIM   64
#define TAU_F   0.5f

typedef __attribute__((ext_vector_type(16))) _Float16 v16h;
typedef __attribute__((ext_vector_type(8)))  _Float16 v8h;
typedef __attribute__((ext_vector_type(8)))  float    v8f;

enum { EPI_NONE = 0, EPI_ELU = 1, EPI_RELU = 2, EPI_TANH = 3 };

__device__ __forceinline__ float elu_f(float x) { return x > 0.f ? x : (expf(x) - 1.f); }

// Fragment K-mapping for 16-bit WMMA operands: element e of the v16h holds
//   k = (e < 8 ? e : 8 + e) + 8*g        (two contiguous 8-half runs)
// where g = lane/16. So a fragment is two 16-byte contiguous loads.

__device__ __forceinline__ v16h load_a_frag(const float* __restrict__ arow, int k0, int g)
{
    const float4 a0 = *(const float4*)(arow + k0 + 8 * g);
    const float4 a1 = *(const float4*)(arow + k0 + 8 * g + 4);
    const float4 a2 = *(const float4*)(arow + k0 + 16 + 8 * g);
    const float4 a3 = *(const float4*)(arow + k0 + 16 + 8 * g + 4);
    v16h af;
    af[0]  = (_Float16)a0.x; af[1]  = (_Float16)a0.y; af[2]  = (_Float16)a0.z; af[3]  = (_Float16)a0.w;
    af[4]  = (_Float16)a1.x; af[5]  = (_Float16)a1.y; af[6]  = (_Float16)a1.z; af[7]  = (_Float16)a1.w;
    af[8]  = (_Float16)a2.x; af[9]  = (_Float16)a2.y; af[10] = (_Float16)a2.z; af[11] = (_Float16)a2.w;
    af[12] = (_Float16)a3.x; af[13] = (_Float16)a3.y; af[14] = (_Float16)a3.z; af[15] = (_Float16)a3.w;
    return af;
}

// ---------------------------------------------------------------------------
// Block GEMM: 128 threads = 4 waves, 128x64 C tile, B staged transposed in LDS.
// C[M,NDIM] = epi( scale * A[M,K] @ B[K,NDIM] + bias )
// Staging: thread (sn, half) loads a contiguous 16-k run of column tn0+sn with
// 16 independent immediate-offset loads (coalesced across sn), converts, and
// writes two ds_store_b128 into the transposed tile. Each wave owns two
// 16-row strips; per K step one staged slab feeds 8 WMMAs, B fragments read
// as ds_load_b128 pairs.
// ---------------------------------------------------------------------------
template <int EPI, bool HAS_BIAS, int NDIM>
__global__ void __launch_bounds__(128)
gemm_block_wmma(const float* __restrict__ A, const float* __restrict__ B,
                const float* __restrict__ bias, float* __restrict__ C,
                int Kdim, float scale)
{
    // Transposed B tile: sB[n][k], row stride 40 halves (80 B, 16B-aligned rows)
    __shared__ __align__(16) _Float16 sB[64][40];

    const int tn0  = blockIdx.x * 64;
    const int tm0  = blockIdx.y * 128;
    const int tid  = threadIdx.x;
    const int wave = tid >> 5;
    const int lane = tid & 31;
    const int g    = lane >> 4;
    const int sub  = lane & 15;
    const int tmA  = tm0 + wave * 16;        // strip 0
    const int tmB  = tm0 + 64 + wave * 16;   // strip 1

    const float* arow0 = A + (size_t)(tmA + sub) * Kdim;
    const float* arow1 = A + (size_t)(tmB + sub) * Kdim;

    // staging coords: thread owns k range [16*half, 16*half+16) of column tn0+sn
    const int sn   = tid & 63;
    const int half = tid >> 6;               // 0..1
    const float* bstage = B + (size_t)(16 * half) * NDIM + tn0 + sn;

    v8f acc0[4] = {};   // strip 0, 4 N sub-tiles
    v8f acc1[4] = {};   // strip 1

    for (int k0 = 0; k0 < Kdim; k0 += 32) {
        {
            float tmp[16];
#pragma unroll
            for (int it = 0; it < 16; ++it)
                tmp[it] = bstage[(size_t)it * NDIM];   // independent, immediate offsets
            v8h pk0, pk1;
#pragma unroll
            for (int i = 0; i < 8; ++i) {
                pk0[i] = (_Float16)tmp[i];
                pk1[i] = (_Float16)tmp[8 + i];
            }
            *(v8h*)&sB[sn][16 * half]     = pk0;       // ds_store_b128
            *(v8h*)&sB[sn][16 * half + 8] = pk1;       // ds_store_b128
        }
        if (k0 + 32 < Kdim) {
            __builtin_prefetch(arow0 + k0 + 32, 0, 1);              // global_prefetch_b8
            __builtin_prefetch(arow1 + k0 + 32, 0, 1);
            __builtin_prefetch(bstage + (size_t)32 * NDIM, 0, 1);
        }
        bstage += (size_t)32 * NDIM;
        __syncthreads();

        const v16h af0 = load_a_frag(arow0, k0, g);
        const v16h af1 = load_a_frag(arow1, k0, g);

#pragma unroll
        for (int nt = 0; nt < 4; ++nt) {
            const _Float16* p = &sB[nt * 16 + sub][g * 8];
            const v8h lo = *(const v8h*)p;          // ds_load_b128
            const v8h hi = *(const v8h*)(p + 16);   // ds_load_b128
            const v16h bf = __builtin_shufflevector(lo, hi,
                    0, 1, 2, 3, 4, 5, 6, 7, 8, 9, 10, 11, 12, 13, 14, 15);
            acc0[nt] = __builtin_amdgcn_wmma_f32_16x16x32_f16(false, af0, false, bf,
                                                              (short)0, acc0[nt], false, false);
            acc1[nt] = __builtin_amdgcn_wmma_f32_16x16x32_f16(false, af1, false, bf,
                                                              (short)0, acc1[nt], false, false);
        }
        __syncthreads();
    }

#pragma unroll
    for (int nt = 0; nt < 4; ++nt) {
        const int n = tn0 + nt * 16 + sub;
        const float bv = HAS_BIAS ? bias[n] : 0.f;
#pragma unroll
        for (int v = 0; v < 8; ++v) {
            const int m0 = tmA + v + (g << 3);
            const int m1 = tmB + v + (g << 3);
            float x0 = acc0[nt][v] * scale + bv;
            float x1 = acc1[nt][v] * scale + bv;
            if constexpr (EPI == EPI_ELU)  { x0 = elu_f(x0);      x1 = elu_f(x1); }
            if constexpr (EPI == EPI_RELU) { x0 = fmaxf(x0, 0.f); x1 = fmaxf(x1, 0.f); }
            if constexpr (EPI == EPI_TANH) { x0 = tanhf(x0);      x1 = tanhf(x1); }
            C[(size_t)m0 * NDIM + n] = x0;
            C[(size_t)m1 * NDIM + n] = x1;
        }
    }
}

// ---------------------------------------------------------------------------
// One-wave GEMM for odd shapes: 16x16 C tile per wave.
// BT=1: C = A[M,K] @ B[N,K]^T (both K-major -> float4 runs).
// BT=0: B is [K,NDIM] with small NDIM (=16 here), immediate-offset loads.
// ---------------------------------------------------------------------------
template <int EPI, bool HAS_BIAS, bool BT, int NDIM>
__global__ void __launch_bounds__(32)
gemm_wave_wmma(const float* __restrict__ A, const float* __restrict__ B,
               const float* __restrict__ bias, float* __restrict__ C,
               int Kdim, float scale)
{
    const int tn   = blockIdx.x * 16;
    const int tm   = blockIdx.y * 16;
    const int lane = threadIdx.x & 31;
    const int g    = lane >> 4;
    const int sub  = lane & 15;

    const float* arow = A + (size_t)(tm + sub) * Kdim;
    const float* brow = B + (size_t)(tn + sub) * Kdim;   // BT path
    const float* bcol = B + tn + sub;                    // !BT path

    v8f acc = {};
    for (int k0 = 0; k0 < Kdim; k0 += 32) {
        if (k0 + 32 < Kdim) {
            __builtin_prefetch(arow + k0 + 32, 0, 1);
            if (BT) __builtin_prefetch(brow + k0 + 32, 0, 1);
        }
        const v16h af = load_a_frag(arow, k0, g);
        v16h bf;
        if constexpr (BT) {
            bf = load_a_frag(brow, k0, g);
        } else {
            const float* bc = bcol + (size_t)k0 * NDIM;
#pragma unroll
            for (int e = 0; e < 16; ++e) {
                const int k = ((e < 8) ? e : (8 + e)) + 8 * g;
                bf[e] = (_Float16)bc[(size_t)k * NDIM];
            }
        }
        acc = __builtin_amdgcn_wmma_f32_16x16x32_f16(false, af, false, bf,
                                                     (short)0, acc, false, false);
    }

    const float bv = HAS_BIAS ? bias[tn + sub] : 0.f;
#pragma unroll
    for (int v = 0; v < 8; ++v) {
        const int m = v + (g << 3);
        float x = acc[v] * scale + bv;
        if constexpr (EPI == EPI_ELU)  x = elu_f(x);
        if constexpr (EPI == EPI_RELU) x = fmaxf(x, 0.f);
        if constexpr (EPI == EPI_TANH) x = tanhf(x);
        C[(size_t)(tm + m) * NDIM + tn + sub] = x;
    }
}

// ---------------------------------------------------------------------------
// Small support kernels
// ---------------------------------------------------------------------------
__global__ void rowinv_kernel(const float* __restrict__ nei, float* __restrict__ inv, int cols)
{
    __shared__ float red[256];
    const int r = blockIdx.x;
    float s = 0.f;
    for (int c = threadIdx.x; c < cols; c += blockDim.x)
        s += nei[(size_t)r * cols + c];
    red[threadIdx.x] = s;
    __syncthreads();
    for (int off = 128; off; off >>= 1) {
        if ((int)threadIdx.x < off) red[threadIdx.x] += red[threadIdx.x + off];
        __syncthreads();
    }
    if (threadIdx.x == 0) inv[r] = 1.f / fmaxf(red[0], 1.f);
}

__global__ void rowscale_kernel(float* __restrict__ x, const float* __restrict__ inv,
                                int count, int cols)
{
    const int i = blockIdx.x * blockDim.x + threadIdx.x;
    if (i < count) x[i] *= inv[i / cols];
}

__global__ void combine_elu_kernel(const float* __restrict__ h_tar,
                                   const float* __restrict__ h_mask,
                                   const float* __restrict__ t,
                                   float* __restrict__ view, float* __restrict__ maskv, int count)
{
    const int i = blockIdx.x * blockDim.x + threadIdx.x;
    if (i < count) {
        const float tv = t[i];
        view[i]  = elu_f(h_tar[i]  + tv);
        maskv[i] = elu_f(h_mask[i] + tv);
    }
}

__global__ void meanadj_kernel(const float* __restrict__ a, const float* __restrict__ b,
                               float* __restrict__ out, int count)
{
    const int i = blockIdx.x * blockDim.x + threadIdx.x;
    if (i < count) out[i] = 0.5f * (a[i] + b[i]);
}

__global__ void l2norm_rows_kernel(float* __restrict__ x, int cols)  // cols == 64
{
    __shared__ float red[64];
    const int r = blockIdx.x;
    float v = x[(size_t)r * cols + threadIdx.x];
    red[threadIdx.x] = v * v;
    __syncthreads();
    for (int off = 32; off; off >>= 1) {
        if ((int)threadIdx.x < off) red[threadIdx.x] += red[threadIdx.x + off];
        __syncthreads();
    }
    const float inv = 1.f / fmaxf(sqrtf(red[0]), 1e-12f);
    x[(size_t)r * cols + threadIdx.x] = v * inv;
}

__global__ void att_reduce_kernel(const float* __restrict__ s, const float* __restrict__ att_vec,
                                  float* __restrict__ e_out, int rows, int cols)
{
    __shared__ float red[256];
    float acc = 0.f;
    const int total = rows * cols;
    for (int i = threadIdx.x; i < total; i += blockDim.x)
        acc += s[i] * att_vec[i & (cols - 1)];
    red[threadIdx.x] = acc;
    __syncthreads();
    for (int off = 128; off; off >>= 1) {
        if ((int)threadIdx.x < off) red[threadIdx.x] += red[threadIdx.x + off];
        __syncthreads();
    }
    if (threadIdx.x == 0) e_out[0] = red[0] / (float)rows;
}

__global__ void softmax_combine_kernel(const float* __restrict__ e,
                                       const float* __restrict__ h0, const float* __restrict__ h1,
                                       const float* __restrict__ h2, const float* __restrict__ h3,
                                       float* __restrict__ z_fine, int count)
{
    const float m  = fmaxf(fmaxf(e[0], e[1]), fmaxf(e[2], e[3]));
    const float w0 = expf(e[0] - m), w1 = expf(e[1] - m);
    const float w2 = expf(e[2] - m), w3 = expf(e[3] - m);
    const float inv = 1.f / (w0 + w1 + w2 + w3);
    const int i = blockIdx.x * blockDim.x + threadIdx.x;
    if (i < count)
        z_fine[i] = (w0 * h0[i] + w1 * h1[i] + w2 * h2[i] + w3 * h3[i]) * inv;
}

// partial[i] = log( sum_j exp(logits[i,j]) * sigmoid(tanh(A1[i]+B1[j]) . w2 + b2) ) - logits[i,i]
__global__ void pair_loss_kernel(const float* __restrict__ logits,
                                 const float* __restrict__ A1, const float* __restrict__ B1,
                                 const float* __restrict__ mlp2_w, const float* __restrict__ mlp2_b,
                                 float* __restrict__ partial, int n)
{
    __shared__ float sA[16];
    __shared__ float sW[16];
    __shared__ float red[256];
    const int i = blockIdx.x;
    if (threadIdx.x < 16) {
        sA[threadIdx.x] = A1[(size_t)i * 16 + threadIdx.x];
        sW[threadIdx.x] = mlp2_w[threadIdx.x];
    }
    __syncthreads();
    const float b2 = mlp2_b[0];
    float denom = 0.f;
    for (int j = threadIdx.x; j < n; j += blockDim.x) {
        float dot = 0.f;
#pragma unroll
        for (int t = 0; t < 16; ++t)
            dot += tanhf(sA[t] + B1[(size_t)j * 16 + t]) * sW[t];
        const float w = 1.f / (1.f + expf(-(dot + b2)));
        denom += expf(logits[(size_t)i * n + j]) * w;
    }
    red[threadIdx.x] = denom;
    __syncthreads();
    for (int off = 128; off; off >>= 1) {
        if ((int)threadIdx.x < off) red[threadIdx.x] += red[threadIdx.x + off];
        __syncthreads();
    }
    if (threadIdx.x == 0)
        partial[i] = logf(red[0]) - logits[(size_t)i * n + i];
}

__global__ void final_mean_kernel(const float* __restrict__ partial, float* __restrict__ out, int n)
{
    __shared__ float red[256];
    float acc = 0.f;
    for (int i = threadIdx.x; i < n; i += blockDim.x) acc += partial[i];
    red[threadIdx.x] = acc;
    __syncthreads();
    for (int off = 128; off; off >>= 1) {
        if ((int)threadIdx.x < off) red[threadIdx.x] += red[threadIdx.x + off];
        __syncthreads();
    }
    if (threadIdx.x == 0) out[0] = red[0] / (float)n;
}

// ---------------------------------------------------------------------------
// Host-side launchers
// ---------------------------------------------------------------------------
template <int EPI, bool HB, int NDIM>
static void block_gemm(const float* A, const float* B, const float* bias, float* C,
                       int M, int K, float scale, hipStream_t s)
{
    dim3 grid(NDIM / 64, M / 128), block(128);
    gemm_block_wmma<EPI, HB, NDIM><<<grid, block, 0, s>>>(A, B, bias, C, K, scale);
}

template <int EPI, bool HB, bool BT, int NDIM>
static void wave_gemm(const float* A, const float* B, const float* bias, float* C,
                      int M, int K, float scale, hipStream_t s)
{
    dim3 grid(NDIM / 16, M / 16), block(32);
    gemm_wave_wmma<EPI, HB, BT, NDIM><<<grid, block, 0, s>>>(A, B, bias, C, K, scale);
}

extern "C" void kernel_launch(void* const* d_in, const int* in_sizes, int n_in,
                              void* d_out, int out_size, void* d_ws, size_t ws_size,
                              hipStream_t stream)
{
    (void)in_sizes; (void)n_in; (void)out_size; (void)ws_size;

    const float* feat0     = (const float*)d_in[0];
    const float* feat1     = (const float*)d_in[1];
    const float* feat2     = (const float*)d_in[2];
    const float* mask_feat = (const float*)d_in[3];
    const float* nei0      = (const float*)d_in[4];
    const float* nei1      = (const float*)d_in[5];
    const float* adj0      = (const float*)d_in[6];
    const float* adj1      = (const float*)d_in[7];
    const float* madj0     = (const float*)d_in[8];
    const float* madj1     = (const float*)d_in[9];
    const float* fc0_w     = (const float*)d_in[10];
    const float* fc0_b     = (const float*)d_in[11];
    const float* fc1_w     = (const float*)d_in[12];
    const float* fc1_b     = (const float*)d_in[13];
    const float* fc2_w     = (const float*)d_in[14];
    const float* fc2_b     = (const float*)d_in[15];
    const float* agg0_w    = (const float*)d_in[16];
    const float* agg1_w    = (const float*)d_in[17];
    const float* gcn_w1    = (const float*)d_in[18];
    const float* gcn_b1    = (const float*)d_in[19];
    const float* gcn_w2    = (const float*)d_in[20];
    const float* gcn_b2    = (const float*)d_in[21];
    const float* att_w     = (const float*)d_in[22];
    const float* att_b     = (const float*)d_in[23];
    const float* att_vec   = (const float*)d_in[24];
    const float* proj_w    = (const float*)d_in[25];
    const float* proj_b    = (const float*)d_in[26];
    const float* mlp1_w    = (const float*)d_in[27];
    const float* mlp1_b    = (const float*)d_in[28];
    const float* mlp2_w    = (const float*)d_in[29];
    const float* mlp2_b    = (const float*)d_in[30];

    // Workspace carve-out
    char* wp = (char*)d_ws;
    auto alloc = [&](size_t nfloats) -> float* {
        float* p = (float*)wp;
        wp += nfloats * sizeof(float);
        return p;
    };
    float* h_tar   = alloc((size_t)N_NODES * H_DIM);
    float* h_mask  = alloc((size_t)N_NODES * H_DIM);
    float* h_nei   = alloc((size_t)M_NODES * H_DIM);
    float* aggbuf  = alloc((size_t)N_NODES * H_DIM);
    float* tbuf    = alloc((size_t)N_NODES * H_DIM);
    float* views0  = alloc((size_t)N_NODES * H_DIM);
    float* masks0  = alloc((size_t)N_NODES * H_DIM);
    float* views1  = alloc((size_t)N_NODES * H_DIM);
    float* masks1  = alloc((size_t)N_NODES * H_DIM);
    float* invcnt  = alloc(N_NODES);
    float* meanA   = alloc((size_t)N_NODES * N_NODES);
    float* xw      = alloc((size_t)N_NODES * E_DIM);
    float* h1buf   = alloc((size_t)N_NODES * E_DIM);
    float* hwbuf   = alloc((size_t)N_NODES * E_DIM);
    float* zco     = alloc((size_t)N_NODES * E_DIM);
    float* hf0     = alloc((size_t)N_NODES * E_DIM);
    float* hf1     = alloc((size_t)N_NODES * E_DIM);
    float* hf2     = alloc((size_t)N_NODES * E_DIM);
    float* hf3     = alloc((size_t)N_NODES * E_DIM);
    float* s_tmp   = alloc((size_t)N_NODES * E_DIM);
    float* e_dev   = alloc(4);
    float* z_fine  = alloc((size_t)N_NODES * E_DIM);
    float* zcp     = alloc((size_t)N_NODES * E_DIM);
    float* zfp     = alloc((size_t)N_NODES * E_DIM);
    float* logits  = alloc((size_t)N_NODES * N_NODES);
    float* A1      = alloc((size_t)N_NODES * 16);
    float* B1      = alloc((size_t)N_NODES * 16);
    float* partial = alloc(N_NODES);

    // --- Stage A: feature encoders + neighbor aggregation ---
    block_gemm<EPI_ELU, true, H_DIM>(feat0,     fc0_w, fc0_b, h_tar,  N_NODES, D0_DIM, 1.f, stream);
    block_gemm<EPI_ELU, true, H_DIM>(mask_feat, fc0_w, fc0_b, h_mask, N_NODES, D0_DIM, 1.f, stream);

    const float* neis[2]   = { nei0, nei1 };
    const float* fcws[2]   = { fc1_w, fc2_w };
    const float* fcbs[2]   = { fc1_b, fc2_b };
    const float* feats[2]  = { feat1, feat2 };
    const float* aggws[2]  = { agg0_w, agg1_w };
    float*       viewsA[2] = { views0, views1 };
    float*       masksA[2] = { masks0, masks1 };

    const int nhCount  = (int)((size_t)N_NODES * H_DIM);
    const int nhBlocks = (nhCount + 255) / 256;
    for (int i = 0; i < 2; ++i) {
        block_gemm<EPI_ELU,  true,  H_DIM>(feats[i], fcws[i], fcbs[i], h_nei, M_NODES, D1_DIM, 1.f, stream);
        block_gemm<EPI_NONE, false, H_DIM>(neis[i],  h_nei,   nullptr, aggbuf, N_NODES, M_NODES, 1.f, stream);
        rowinv_kernel<<<N_NODES, 256, 0, stream>>>(neis[i], invcnt, M_NODES);
        rowscale_kernel<<<nhBlocks, 256, 0, stream>>>(aggbuf, invcnt, nhCount, H_DIM);
        block_gemm<EPI_NONE, false, H_DIM>(aggbuf, aggws[i], nullptr, tbuf, N_NODES, H_DIM, 1.f, stream);
        combine_elu_kernel<<<nhBlocks, 256, 0, stream>>>(h_tar, h_mask, tbuf,
                                                         viewsA[i], masksA[i], nhCount);
    }

    // --- Stage B: GCNs ---
    const int nnCount  = (int)((size_t)N_NODES * N_NODES);
    const int nnBlocks = (nnCount + 255) / 256;
    meanadj_kernel<<<nnBlocks, 256, 0, stream>>>(adj0, adj1, meanA, nnCount);

    auto run_gcn = [&](const float* x, int xK, const float* adj, float* out) {
        block_gemm<EPI_NONE, false, E_DIM>(x,     gcn_w1, nullptr, xw,    N_NODES, xK,      1.f, stream);
        block_gemm<EPI_RELU, true,  E_DIM>(adj,   xw,     gcn_b1,  h1buf, N_NODES, N_NODES, 1.f, stream);
        block_gemm<EPI_NONE, false, E_DIM>(h1buf, gcn_w2, nullptr, hwbuf, N_NODES, E_DIM,   1.f, stream);
        block_gemm<EPI_NONE, true,  E_DIM>(adj,   hwbuf,  gcn_b2,  out,   N_NODES, N_NODES, 1.f, stream);
    };
    run_gcn(h_tar,  H_DIM, meanA, zco);
    run_gcn(views0, H_DIM, adj0,  hf0);
    run_gcn(masks0, H_DIM, madj0, hf1);
    run_gcn(views1, H_DIM, adj1,  hf2);
    run_gcn(masks1, H_DIM, madj1, hf3);

    // --- Stage C: semantic attention ---
    float* hfv[4] = { hf0, hf1, hf2, hf3 };
    for (int v = 0; v < 4; ++v)
        l2norm_rows_kernel<<<N_NODES, 64, 0, stream>>>(hfv[v], E_DIM);
    for (int v = 0; v < 4; ++v) {
        block_gemm<EPI_TANH, true, E_DIM>(hfv[v], att_w, att_b, s_tmp, N_NODES, E_DIM, 1.f, stream);
        att_reduce_kernel<<<1, 256, 0, stream>>>(s_tmp, att_vec, e_dev + v, N_NODES, E_DIM);
    }
    const int neCount  = (int)((size_t)N_NODES * E_DIM);
    const int neBlocks = (neCount + 255) / 256;
    softmax_combine_kernel<<<neBlocks, 256, 0, stream>>>(e_dev, hf0, hf1, hf2, hf3,
                                                         z_fine, neCount);

    // --- Stage D: projections + weighted InfoNCE ---
    block_gemm<EPI_TANH, true, E_DIM>(zco,    proj_w, proj_b, zcp, N_NODES, E_DIM, 1.f, stream);
    block_gemm<EPI_TANH, true, E_DIM>(z_fine, proj_w, proj_b, zfp, N_NODES, E_DIM, 1.f, stream);
    l2norm_rows_kernel<<<N_NODES, 64, 0, stream>>>(zcp, E_DIM);
    l2norm_rows_kernel<<<N_NODES, 64, 0, stream>>>(zfp, E_DIM);

    // logits = (zf @ zc^T) / TAU   (B transposed, K = E)
    wave_gemm<EPI_NONE, false, true, N_NODES>(zfp, zcp, nullptr, logits,
                                              N_NODES, E_DIM, 1.f / TAU_F, stream);

    // Pair-MLP factorization: A1 = zf@W1 + b1, B1 = zc@W1  (N = 16)
    wave_gemm<EPI_NONE, true,  false, 16>(zfp, mlp1_w, mlp1_b,  A1, N_NODES, E_DIM, 1.f, stream);
    wave_gemm<EPI_NONE, false, false, 16>(zcp, mlp1_w, nullptr, B1, N_NODES, E_DIM, 1.f, stream);

    pair_loss_kernel<<<N_NODES, 256, 0, stream>>>(logits, A1, B1, mlp2_w, mlp2_b,
                                                  partial, N_NODES);
    final_mean_kernel<<<1, 256, 0, stream>>>(partial, (float*)d_out, N_NODES);
}
